// GCN_37838661878509
// MI455X (gfx1250) — matrix operand order repaired and checked by reading, not compiled, not run on hardware
//
#include <hip/hip_runtime.h>
#include <hip/hip_bf16.h>

// ---------------------------------------------------------------------------
// GCN 2-layer forward for MI455X (gfx1250, wave32).
//   GEMMs: v_wmma_f32_16x16x32_bf16, weights pre-packed into B-fragment
//   layout, staged in LDS via global_load_async_to_lds_b128 (ASYNCcnt),
//   read back with ds_load_b128. Scatter: native f32 atomics (L2-resident).
// ---------------------------------------------------------------------------

typedef __attribute__((ext_vector_type(16))) __bf16          v16bf;
typedef __attribute__((ext_vector_type(8)))  float           v8f;
typedef __attribute__((ext_vector_type(8)))  unsigned short  u16x8;

union FragBF {
    v16bf          bf;
    u16x8          h[2];
    unsigned short us[16];
};

__device__ __forceinline__ unsigned short f32_to_bf16_rne(float f) {
    unsigned int u = __float_as_uint(f);
    u += 0x7FFFu + ((u >> 16) & 1u);   // round-to-nearest-even
    return (unsigned short)(u >> 16);
}

// ---------------------------------------------------------------------------
// Degree / normalization
// ---------------------------------------------------------------------------
__global__ void gcn_init_deg(float* __restrict__ deg, int n) {
    int i = blockIdx.x * blockDim.x + threadIdx.x;
    if (i < n) deg[i] = 1.0f;                       // self-loop contribution
}

__global__ void gcn_deg_accum(const int* __restrict__ dst, int E,
                              float* __restrict__ deg) {
    int e = blockIdx.x * blockDim.x + threadIdx.x;
    if (e < E) unsafeAtomicAdd(&deg[dst[e]], 1.0f); // global_atomic_add_f32
}

__global__ void gcn_deg_to_dinv(float* __restrict__ deg, int n) {
    int i = blockIdx.x * blockDim.x + threadIdx.x;
    if (i < n) {
        float d = deg[i];
        deg[i] = (d > 0.0f) ? rsqrtf(d) : 0.0f;
    }
}

// ---------------------------------------------------------------------------
// fp32 -> bf16 conversion (A operands)
// ---------------------------------------------------------------------------
__global__ void gcn_f32_to_bf16(const float* __restrict__ in,
                                unsigned short* __restrict__ out, size_t n) {
    size_t i = (size_t)blockIdx.x * blockDim.x + threadIdx.x;
    if (i < n) out[i] = f32_to_bf16_rne(in[i]);
}

// out1 + b -> relu -> bf16 (feeds GEMM of layer 2)
__global__ void gcn_bias_relu_bf16(const float* __restrict__ in,
                                   const float* __restrict__ bias,
                                   unsigned short* __restrict__ out,
                                   size_t total, int C) {
    size_t i = (size_t)blockIdx.x * blockDim.x + threadIdx.x;
    if (i < total) {
        int   c = (int)(i % (size_t)C);
        float v = in[i] + bias[c];
        out[i]  = f32_to_bf16_rne(v > 0.0f ? v : 0.0f);
    }
}

// ---------------------------------------------------------------------------
// Pack W[K,OC] (fp32, row-major) into WMMA B-fragment layout (bf16):
//   Bp[((ct*KSTEPS + ks)*32 + lane)*16 + j]
//     = bf16( W[ ks*32 + (lane>>4)*16 + j ][ ct*16 + (lane&15) ] )
// ---------------------------------------------------------------------------
__global__ void gcn_pack_B(const float* __restrict__ W,
                           unsigned short* __restrict__ Bp,
                           int K, int OC) {
    int idx = blockIdx.x * blockDim.x + threadIdx.x;
    if (idx >= K * OC) return;
    const int ksteps = K >> 5;
    int j    = idx & 15;
    int lane = (idx >> 4) & 31;
    int rest = idx >> 9;                 // ct*ksteps + ks
    int ct   = rest / ksteps;
    int ks   = rest - ct * ksteps;
    int k    = ks * 32 + (lane >> 4) * 16 + j;
    int n    = ct * 16 + (lane & 15);
    Bp[idx]  = f32_to_bf16_rne(W[(size_t)k * OC + n]);
}

// ---------------------------------------------------------------------------
// WMMA GEMM: H[N,OC] = A[N,K](bf16) @ W(packed bf16), fp32 accumulate.
// Grid: (rowBlocks, colGroups). Block = 8 waves; each wave owns one 16-row
// tile x the block's 4 column tiles. Packed B for the col group (16 KB) is
// staged into LDS once per block with async-to-LDS loads, then every WMMA
// reads its B fragment from LDS (immediate-offset ds_load_b128).
// K = KSTEPS*32 (=128 here); OC multiple of 64.
// ---------------------------------------------------------------------------
template <int KSTEPS>
__global__ __launch_bounds__(256)
void gcn_gemm_bf16(const unsigned short* __restrict__ A,
                   const unsigned short* __restrict__ Bp,
                   float* __restrict__ H,
                   int N, int OC) {
    constexpr int K          = KSTEPS * 32;
    constexpr int SEG_ELEMS  = 4 * KSTEPS * 512;       // 4 col tiles of frags
    constexpr int SEG_BYTES  = SEG_ELEMS * 2;          // 16 KB for KSTEPS=4

    __shared__ unsigned short sB[SEG_ELEMS];

    const int tid      = threadIdx.x;
    const int wave     = tid >> 5;
    const int lane     = tid & 31;
    const int colGroup = blockIdx.y;

    // ---- stage packed B for this col group into LDS (async, ASYNCcnt) ----
    {
        const char* gsrc = (const char*)(Bp + (size_t)colGroup * SEG_ELEMS);
        unsigned ldsBase = (unsigned)(size_t)(&sB[0]);   // low 32 bits = LDS offset
#pragma unroll
        for (int it = 0; it < SEG_BYTES / 4096; ++it) {
            unsigned    ldsOff = ldsBase + it * 4096 + tid * 16;
            const char* g      = gsrc + it * 4096 + tid * 16;
            asm volatile("global_load_async_to_lds_b128 %0, %1, off"
                         :: "v"(ldsOff), "v"(g) : "memory");
        }
        asm volatile("s_wait_asynccnt 0x0" ::: "memory");
    }
    __syncthreads();

    const int rowTiles = (N + 15) >> 4;
    const int rowTile  = blockIdx.x * 8 + wave;
    if (rowTile >= rowTiles) return;                 // wave-uniform, after barrier

    const int hi = lane >> 4;
    const int lo = lane & 15;

    int m = (rowTile << 4) + lo;
    if (m >= N) m = N - 1;                           // clamp (N=50000 is exact)
    const unsigned short* Arow = A + (size_t)m * K + hi * 16;
    const unsigned short* sLane = sB + lane * 16;    // per-lane base in LDS

    v8f acc[4] = {};
#pragma unroll
    for (int ks = 0; ks < KSTEPS; ++ks) {
        FragBF a;
        a.h[0] = *(const u16x8*)(Arow + ks * 32);        // 16B global
        a.h[1] = *(const u16x8*)(Arow + ks * 32 + 8);    // 16B global
#pragma unroll
        for (int c = 0; c < 4; ++c) {
            const unsigned short* sp = sLane + (c * KSTEPS + ks) * 512;
            FragBF b;
            b.h[0] = *(const u16x8*)sp;                  // ds_load_b128
            b.h[1] = *(const u16x8*)(sp + 8);
            acc[c] = __builtin_amdgcn_wmma_f32_16x16x32_bf16(
                         false, a.bf, false, b.bf, (short)0, acc[c],
                         false, false);
        }
    }

    // D layout: VGPR i -> row (rowTile*16 + hi*8 + i), col (ct*16 + lo)
    const int mbase = (rowTile << 4) + hi * 8;
#pragma unroll
    for (int c = 0; c < 4; ++c) {
        const int n = ((colGroup * 4 + c) << 4) + lo;
#pragma unroll
        for (int i = 0; i < 8; ++i) {
            int r = mbase + i;
            if (r < N) H[(size_t)r * OC + n] = acc[c][i];
        }
    }
}

// ---------------------------------------------------------------------------
// Self-loop term (plain write, establishes out): out = h * dinv^2 [+ bias]
// ---------------------------------------------------------------------------
__global__ void gcn_self(const float* __restrict__ h,
                         const float* __restrict__ dinv,
                         const float* __restrict__ bias,   // may be nullptr
                         float* __restrict__ out,
                         size_t total, int C) {
    size_t i = (size_t)blockIdx.x * blockDim.x + threadIdx.x;
    if (i < total) {
        size_t node = i / (size_t)C;
        int    c    = (int)(i - node * (size_t)C);
        float  dv   = dinv[node];
        float  v    = h[i] * dv * dv;
        if (bias) v += bias[c];
        out[i] = v;
    }
}

// ---------------------------------------------------------------------------
// Edge scatter: one wave per edge; lane handles C/32 contiguous features.
// Coalesced 128B gather of h[src], native f32 atomics into out[dst].
// ---------------------------------------------------------------------------
__global__ __launch_bounds__(256)
void gcn_edge(const float* __restrict__ h,
              const float* __restrict__ dinv,
              const int* __restrict__ src,
              const int* __restrict__ dst,
              float* __restrict__ out,
              int E, int C) {
    const int wavesPerBlock = blockDim.x >> 5;
    const int e    = blockIdx.x * wavesPerBlock + (threadIdx.x >> 5);
    if (e >= E) return;
    const int lane = threadIdx.x & 31;
    const int per  = C >> 5;                       // 4 (C=128) or 2 (C=64)

    const int   s   = src[e];
    const int   d   = dst[e];
    const float nrm = dinv[s] * dinv[d];

    const float* hs = h   + (size_t)s * C + lane * per;
    float*       od = out + (size_t)d * C + lane * per;
#pragma unroll 4
    for (int j = 0; j < per; ++j)
        unsafeAtomicAdd(&od[j], hs[j] * nrm);      // global_atomic_add_f32
}

// ---------------------------------------------------------------------------
// Launch
// ---------------------------------------------------------------------------
extern "C" void kernel_launch(void* const* d_in, const int* in_sizes, int n_in,
                              void* d_out, int out_size, void* d_ws, size_t ws_size,
                              hipStream_t stream) {
    const float* x  = (const float*)d_in[0];
    const int*   ei = (const int*)d_in[1];   // edge_index flattened [2, E]
    const float* W1 = (const float*)d_in[2];
    const float* b1 = (const float*)d_in[3];
    const float* W2 = (const float*)d_in[4];
    const float* b2 = (const float*)d_in[5];

    const int IN_C = 128, HID_C = 128, OUT_C = 64;
    const int N = in_sizes[0] / IN_C;
    const int E = in_sizes[1] / 2;
    const int* src = ei;
    const int* dst = ei + E;

    // Workspace carve-up (256B aligned). Total ~77 MB.
    char*  ws  = (char*)d_ws;
    size_t off = 0;
    auto carve = [&](size_t bytes) -> char* {
        char* p = ws + off;
        off = (off + bytes + 255) & ~(size_t)255;
        return p;
    };
    float*          dinv = (float*)         carve((size_t)N * 4);
    unsigned short* xbf  = (unsigned short*)carve((size_t)N * IN_C * 2);  // reused as h1act
    unsigned short* w1p  = (unsigned short*)carve((size_t)IN_C * HID_C * 2);
    unsigned short* w2p  = (unsigned short*)carve((size_t)HID_C * OUT_C * 2);
    float*          h1   = (float*)         carve((size_t)N * HID_C * 4);
    float*          out1 = (float*)         carve((size_t)N * HID_C * 4);
    float*          h2   = (float*)         carve((size_t)N * OUT_C * 4);
    float*          out  = (float*)d_out;

    const int T = 256;
    const size_t n1 = (size_t)N * HID_C;   // 6.4M
    const size_t n2 = (size_t)N * OUT_C;   // 3.2M

    // --- normalization ---
    gcn_init_deg   <<<(N + T - 1) / T, T, 0, stream>>>(dinv, N);
    gcn_deg_accum  <<<(E + T - 1) / T, T, 0, stream>>>(dst, E, dinv);
    gcn_deg_to_dinv<<<(N + T - 1) / T, T, 0, stream>>>(dinv, N);

    // --- operand prep: x -> bf16, W -> packed bf16 B-fragments ---
    gcn_f32_to_bf16<<<(int)((n1 + T - 1) / T), T, 0, stream>>>(x, xbf, n1);
    gcn_pack_B<<<(IN_C * HID_C + T - 1) / T, T, 0, stream>>>(W1, w1p, IN_C, HID_C);
    gcn_pack_B<<<(HID_C * OUT_C + T - 1) / T, T, 0, stream>>>(W2, w2p, HID_C, OUT_C);

    const int rowTiles  = (N + 15) / 16;
    const int rowBlocks = (rowTiles + 7) / 8;

    // --- layer 1: h1 = x @ W1 ; out1 = scatter(norm * h1) ; act -> bf16 ---
    gcn_gemm_bf16<4><<<dim3(rowBlocks, HID_C / 64), T, 0, stream>>>(xbf, w1p, h1, N, HID_C);
    gcn_self<<<(int)((n1 + T - 1) / T), T, 0, stream>>>(h1, dinv, nullptr, out1, n1, HID_C);
    gcn_edge<<<(E + 7) / 8, T, 0, stream>>>(h1, dinv, src, dst, out1, E, HID_C);
    gcn_bias_relu_bf16<<<(int)((n1 + T - 1) / T), T, 0, stream>>>(out1, b1, xbf, n1, HID_C);

    // --- layer 2: h2 = act @ W2 ; out = scatter(norm * h2) + b2 ---
    gcn_gemm_bf16<4><<<dim3(rowBlocks, OUT_C / 64), T, 0, stream>>>(xbf, w2p, h2, N, OUT_C);
    gcn_self<<<(int)((n2 + T - 1) / T), T, 0, stream>>>(h2, dinv, b2, out, n2, OUT_C);
    gcn_edge<<<(E + 7) / 8, T, 0, stream>>>(h2, dinv, src, dst, out, E, OUT_C);
}
